// TransformerWithLocalAttention_24343874634038
// MI455X (gfx1250) — matrix operand order repaired and checked by reading, not compiled, not run on hardware
//
#include <hip/hip_runtime.h>

typedef __attribute__((ext_vector_type(16))) __bf16 v16bf;
typedef __attribute__((ext_vector_type(8)))  __bf16 v8bf;
typedef __attribute__((ext_vector_type(8)))  float  v8f;
typedef unsigned short u16;

// ---------- helpers ----------

__device__ __forceinline__ u16 f2bf(float f) {
  union { float f; unsigned u; } v; v.f = f;
  unsigned u = v.u;
  return (u16)((u + 0x7FFFu + ((u >> 16) & 1u)) >> 16);   // RNE
}

// A-matrix (16x32 bf16) fragment: lane m=lane%16, half=lane/16.
// halves i<8 -> K = koff+i ; i>=8 -> K = koff+16+(i-8), with koff = kk + 8*half.
__device__ __forceinline__ v16bf frag_a(const u16* p, int koff) {
  v8bf lo = *(const v8bf*)(p + koff);
  v8bf hi = *(const v8bf*)(p + koff + 16);
  v16bf r;
#pragma unroll
  for (int i = 0; i < 8; ++i) { r[i] = lo[i]; r[i + 8] = hi[i]; }
  return r;
}

// B-matrix (32x16 bf16) fragment: lane n=lane%16,
// halves i<8 -> K = koff+i ; i>=8 -> K = koff+16+(i-8), with koff = kk + 8*half.
__device__ __forceinline__ v16bf frag_b(const u16* p, int koff) {
  v8bf lo = *(const v8bf*)(p + koff);
  v8bf hi = *(const v8bf*)(p + koff + 16);
  v16bf r;
#pragma unroll
  for (int i = 0; i < 8; ++i) { r[i] = lo[i]; r[i + 8] = hi[i]; }
  return r;
}

__device__ __forceinline__ v8f wmma_bf16(v16bf a, v16bf b, v8f c) {
  return __builtin_amdgcn_wmma_f32_16x16x32_bf16(false, a, false, b,
                                                 (short)0, c, false, false);
}

// ---------- prep kernels ----------

__global__ void cast_bf16_kernel(const float* __restrict__ in,
                                 u16* __restrict__ out, size_t n) {
  size_t i = (size_t)blockIdx.x * 256 + threadIdx.x;
  if (i < n) out[i] = f2bf(in[i]);
}

// in[R][C] (fp32) -> out[C][R] (bf16)
__global__ void transpose_cast_kernel(const float* __restrict__ in,
                                      u16* __restrict__ out, int R, int C) {
  size_t i = (size_t)blockIdx.x * 256 + threadIdx.x;
  if (i >= (size_t)R * C) return;
  int r = (int)(i % R), c = (int)(i / R);
  out[(size_t)c * R + r] = f2bf(in[(size_t)r * C + c]);
}

// ---------- generic bf16 WMMA GEMM:  Y = act(A @ Bt^T + bias) ----------
// A:[M][K] bf16 row-major, Bt:[N][K] bf16 (pre-transposed weights).
// Workgroup tile 128x128: 8 waves, each wave 32(M) x 64(N) = 2x4 WMMA tiles.

__global__ __launch_bounds__(256)
void gemm_bf16_kernel(const u16* __restrict__ A, const u16* __restrict__ Bt,
                      const float* __restrict__ bias,
                      float* __restrict__ outF, u16* __restrict__ outB,
                      int M, int N, int K, int relu) {
  const int tid  = threadIdx.x;
  const int lane = tid & 31;
  const int w    = tid >> 5;           // 0..7
  const int wm   = w & 3, wn = w >> 2; // 4 waves in M, 2 waves in N
  const int m16  = lane & 15, half = lane >> 4;

  const int rowBase = blockIdx.y * 128 + wm * 32;
  const int colBase = blockIdx.x * 128 + wn * 64;

  v8f acc[2][4] = {};

  const u16* a0 = A + (size_t)(rowBase + m16) * K;
  const u16* a1 = A + (size_t)(rowBase + 16 + m16) * K;
  const u16* bp[4];
#pragma unroll
  for (int tj = 0; tj < 4; ++tj)
    bp[tj] = Bt + (size_t)(colBase + tj * 16 + m16) * K;

  for (int kk = 0; kk < K; kk += 32) {
    __builtin_prefetch(a0 + kk + 128, 0, 1);
    __builtin_prefetch(bp[0] + kk + 128, 0, 1);
    const int ka = kk + 8 * half;
    const int kb = kk + 8 * half;
    v16bf fa0 = frag_a(a0, ka);
    v16bf fa1 = frag_a(a1, ka);
#pragma unroll
    for (int tj = 0; tj < 4; ++tj) {
      v16bf fb = frag_b(bp[tj], kb);
      acc[0][tj] = wmma_bf16(fa0, fb, acc[0][tj]);
      acc[1][tj] = wmma_bf16(fa1, fb, acc[1][tj]);
    }
  }

#pragma unroll
  for (int ti = 0; ti < 2; ++ti)
#pragma unroll
    for (int tj = 0; tj < 4; ++tj) {
      const int col = colBase + tj * 16 + m16;
      const float bv = bias ? bias[col] : 0.f;
#pragma unroll
      for (int r = 0; r < 8; ++r) {
        const int row = rowBase + ti * 16 + half * 8 + r;
        float v = acc[ti][tj][r] + bv;
        if (relu) v = v > 0.f ? v : 0.f;
        const size_t o = (size_t)row * N + col;
        if (outF) outF[o] = v;
        if (outB) outB[o] = f2bf(v);
      }
    }
}

// ---------- block-local attention ----------
// One workgroup per (qblock, batch). 64 queries x 192-key window, D=512.

#define SSTR 193   // fp32 scores stride (bank-conflict pad)
#define PSTR 208   // bf16 stride, 416B = 26*16 -> 16B-aligned ds_read_b128

__global__ __launch_bounds__(256)
void attn_kernel(const u16* __restrict__ Q, const u16* __restrict__ Kb,
                 const u16* __restrict__ Vb, float* __restrict__ outF) {
  __shared__ float S[64 * SSTR];        // 49.4 KB
  __shared__ u16   P[64 * PSTR];        // 26.6 KB
  __shared__ u16   Vt[128 * PSTR];      // 53.2 KB

  const int qb = blockIdx.x, b = blockIdx.y;
  const int tid = threadIdx.x, lane = tid & 31, w = tid >> 5;
  const int m16 = lane & 15, half = lane >> 4;
  const float scale = 0.04419417382415922f;  // 1/sqrt(512)
  const size_t baseRow = (size_t)b * 4096 + (size_t)qb * 64;

  // ---- phase 1: S = Q Kwin^T * scale (48 tiles of 16x16, 6 per wave) ----
  for (int t = w * 6; t < w * 6 + 6; ++t) {
    const int mi = t / 12, ni = t % 12;
    const int kblk = qb - 1 + ni / 4;
    const int colLocal = ni * 16 + m16;
    if (kblk < 0 || kblk >= 64) {
#pragma unroll
      for (int r = 0; r < 8; ++r)
        S[(mi * 16 + half * 8 + r) * SSTR + colLocal] = -1e9f;
      continue;
    }
    const u16* qp = Q + (baseRow + mi * 16 + m16) * 512;
    const int wkey = ni * 16 + m16;
    const u16* kp = Kb + ((size_t)b * 4096 + (size_t)kblk * 64 + (wkey & 63)) * 512;
    v8f acc = {};
    for (int kk = 0; kk < 512; kk += 32) {
      v16bf fa = frag_a(qp, kk + 8 * half);
      v16bf fb = frag_b(kp, kk + 8 * half);   // K rows are K-contiguous = B columns
      acc = wmma_bf16(fa, fb, acc);
    }
#pragma unroll
    for (int r = 0; r < 8; ++r)
      S[(mi * 16 + half * 8 + r) * SSTR + colLocal] = acc[r] * scale;
  }
  __syncthreads();

  // ---- phase 2: softmax rows, write P (bf16) ----
  if (tid < 64) {
    const int row = tid;
    float mmax = -3.4e38f;
    for (int c = 0; c < 192; ++c) mmax = fmaxf(mmax, S[row * SSTR + c]);
    float sum = 0.f;
    for (int c = 0; c < 192; ++c) sum += __expf(S[row * SSTR + c] - mmax);
    const float inv = 1.f / sum;
    for (int c = 0; c < 192; ++c)
      P[row * PSTR + c] = f2bf(__expf(S[row * SSTR + c] - mmax) * inv);
  }
  __syncthreads();

  // ---- phase 3: O = P @ Vwin, D processed in 4 chunks of 128 ----
  for (int dc = 0; dc < 4; ++dc) {
    // stage Vt[dd][key] = V[key][dc*128+dd]  (clamped blocks contribute ~0)
    for (int idx = tid; idx < 192 * 128; idx += 256) {
      const int dd = idx & 127, key = idx >> 7;
      int kblk = qb - 1 + (key >> 6);
      kblk = kblk < 0 ? 0 : (kblk > 63 ? 63 : kblk);
      Vt[dd * PSTR + key] =
          Vb[((size_t)b * 4096 + (size_t)kblk * 64 + (key & 63)) * 512 + dc * 128 + dd];
    }
    __syncthreads();

    const int mi = w & 3;
    const int nj0 = (w >> 2) * 4;
    v8f acc[4] = {};
    const int prow = mi * 16 + m16;
    for (int kk = 0; kk < 192; kk += 32) {
      v16bf fa = frag_a(&P[prow * PSTR], kk + 8 * half);
#pragma unroll
      for (int j = 0; j < 4; ++j) {
        const int dd = (nj0 + j) * 16 + m16;
        v16bf fb = frag_b(&Vt[dd * PSTR], kk + 8 * half);
        acc[j] = wmma_bf16(fa, fb, acc[j]);
      }
    }
#pragma unroll
    for (int j = 0; j < 4; ++j) {
      const int col = dc * 128 + (nj0 + j) * 16 + m16;
#pragma unroll
      for (int r = 0; r < 8; ++r) {
        const size_t row = baseRow + mi * 16 + half * 8 + r;
        outF[row * 512 + col] = acc[j][r];
      }
    }
    __syncthreads();
  }
}

// ---------- residual + LayerNorm (D=512, 128 threads x 4 elems) ----------

__global__ __launch_bounds__(128)
void add_ln_kernel(const float* __restrict__ X, const float* __restrict__ Y,
                   const float* __restrict__ g, const float* __restrict__ be,
                   float* __restrict__ outF, u16* __restrict__ outB) {
  __shared__ float red[128];
  const int tid = threadIdx.x;
  const size_t base = (size_t)blockIdx.x * 512;

  float v[4];
#pragma unroll
  for (int i = 0; i < 4; ++i) {
    const int c = tid + i * 128;
    v[i] = X[base + c] + Y[base + c];
  }
  float s = v[0] + v[1] + v[2] + v[3];
  red[tid] = s; __syncthreads();
  for (int off = 64; off > 0; off >>= 1) {
    if (tid < off) red[tid] += red[tid + off];
    __syncthreads();
  }
  const float mean = red[0] * (1.f / 512.f);
  __syncthreads();

  float s2 = 0.f;
#pragma unroll
  for (int i = 0; i < 4; ++i) { const float d = v[i] - mean; s2 += d * d; }
  red[tid] = s2; __syncthreads();
  for (int off = 64; off > 0; off >>= 1) {
    if (tid < off) red[tid] += red[tid + off];
    __syncthreads();
  }
  const float inv = rsqrtf(red[0] * (1.f / 512.f) + 1e-6f);

#pragma unroll
  for (int i = 0; i < 4; ++i) {
    const int c = tid + i * 128;
    const float o = g[c] * (v[i] - mean) * inv + be[c];
    if (outF) outF[base + c] = o;
    if (outB) outB[base + c] = f2bf(o);
  }
}

// ---------- host launcher ----------

extern "C" void kernel_launch(void* const* d_in, const int* in_sizes, int n_in,
                              void* d_out, int out_size, void* d_ws, size_t ws_size,
                              hipStream_t stream) {
  const float* x   = (const float*)d_in[0];
  const float* Wq  = (const float*)d_in[1];
  const float* bq  = (const float*)d_in[2];
  const float* Wk  = (const float*)d_in[3];
  const float* bk  = (const float*)d_in[4];
  const float* Wv  = (const float*)d_in[5];
  const float* bv  = (const float*)d_in[6];
  const float* W1  = (const float*)d_in[7];
  const float* b1  = (const float*)d_in[8];
  const float* W2  = (const float*)d_in[9];
  const float* b2  = (const float*)d_in[10];
  const float* g1  = (const float*)d_in[11];
  const float* be1 = (const float*)d_in[12];
  const float* g2  = (const float*)d_in[13];
  const float* be2 = (const float*)d_in[14];

  char* ws = (char*)d_ws;
  const size_t NTOK = 32768;                 // B*S
  const size_t BF_TOKD = NTOK * 512 * 2;     // 32 MB
  const size_t F32_TOKD = NTOK * 512 * 4;    // 64 MB

  u16*   xbf   = (u16*)(ws);
  u16*   Qbf   = (u16*)(ws + 1 * BF_TOKD);
  u16*   Kbf   = (u16*)(ws + 2 * BF_TOKD);
  u16*   Vbf   = (u16*)(ws + 3 * BF_TOKD);
  u16*   Hbf   = (u16*)(ws);                 // 128 MB, aliases xbf|Q|K|V (dead then)
  float* attnF = (float*)(ws + 4 * BF_TOKD); // reused as FFN2 output
  float* out1F = (float*)(ws + 4 * BF_TOKD + F32_TOKD);
  u16*   out1B = (u16*)(ws + 4 * BF_TOKD + 2 * F32_TOKD);
  char*  wb    = ws + 4 * BF_TOKD + 2 * F32_TOKD + BF_TOKD;
  u16* WqT = (u16*)wb;
  u16* WkT = WqT + 512 * 512;
  u16* WvT = WkT + 512 * 512;
  u16* W1T = WvT + 512 * 512;                // [2048][512]
  u16* W2T = W1T + 512 * 2048;               // [512][2048]

  // prep
  cast_bf16_kernel<<<(unsigned)((NTOK * 512) / 256), 256, 0, stream>>>(x, xbf, NTOK * 512);
  transpose_cast_kernel<<<(512 * 512) / 256, 256, 0, stream>>>(Wq, WqT, 512, 512);
  transpose_cast_kernel<<<(512 * 512) / 256, 256, 0, stream>>>(Wk, WkT, 512, 512);
  transpose_cast_kernel<<<(512 * 512) / 256, 256, 0, stream>>>(Wv, WvT, 512, 512);
  transpose_cast_kernel<<<(512 * 2048) / 256, 256, 0, stream>>>(W1, W1T, 512, 2048);
  transpose_cast_kernel<<<(2048 * 512) / 256, 256, 0, stream>>>(W2, W2T, 2048, 512);

  // QKV projections (grid: N/128 x M/128)
  dim3 gP(512 / 128, (unsigned)(NTOK / 128));
  gemm_bf16_kernel<<<gP, 256, 0, stream>>>(xbf, WqT, bq, nullptr, Qbf, (int)NTOK, 512, 512, 0);
  gemm_bf16_kernel<<<gP, 256, 0, stream>>>(xbf, WkT, bk, nullptr, Kbf, (int)NTOK, 512, 512, 0);
  gemm_bf16_kernel<<<gP, 256, 0, stream>>>(xbf, WvT, bv, nullptr, Vbf, (int)NTOK, 512, 512, 0);

  // local attention
  attn_kernel<<<dim3(64, 8), 256, 0, stream>>>(Qbf, Kbf, Vbf, attnF);

  // LN1
  add_ln_kernel<<<(unsigned)NTOK, 128, 0, stream>>>(x, attnF, g1, be1, out1F, out1B);

  // FFN
  gemm_bf16_kernel<<<dim3(2048 / 128, (unsigned)(NTOK / 128)), 256, 0, stream>>>(
      out1B, W1T, b1, nullptr, Hbf, (int)NTOK, 2048, 512, 1);
  gemm_bf16_kernel<<<dim3(512 / 128, (unsigned)(NTOK / 128)), 256, 0, stream>>>(
      Hbf, W2T, b2, attnF, nullptr, (int)NTOK, 512, 2048, 0);

  // LN2 -> final output
  add_ln_kernel<<<(unsigned)NTOK, 128, 0, stream>>>(out1F, attnF, g2, be2,
                                                    (float*)d_out, nullptr);
}
